// MultiHeadAttention_56075093017289
// MI455X (gfx1250) — matrix operand order repaired and checked by reading, not compiled, not run on hardware
//
#include <hip/hip_runtime.h>

typedef __bf16 bf16;
typedef __attribute__((ext_vector_type(16))) __bf16 v16bf;
typedef __attribute__((ext_vector_type(8)))  __bf16 v8bf;
typedef __attribute__((ext_vector_type(8)))  float  v8f;
typedef __attribute__((ext_vector_type(4)))  float  v4f;

#define EMBED 256
#define HEADS 8
#define GROUP 64
#define HD    32
#define SM_SCALE 0.0625f   // 1/sqrt(256)

// ---------- WMMA helpers -----------------------------------------------------
__device__ __forceinline__ v8f wmma_bf16(v16bf a, v16bf b, v8f c) {
  return __builtin_amdgcn_wmma_f32_16x16x32_bf16(false, a, false, b,
                                                 (short)0, c, false, false);
}
__device__ __forceinline__ v8f zero8() {
  v8f z = {0.f, 0.f, 0.f, 0.f, 0.f, 0.f, 0.f, 0.f};
  return z;
}
// A operand (16x32 bf16): lane holds K = kbase..kbase+7 and 16+kbase..16+kbase+7
// (kbase = 8*(lane/16)); caller passes &M[row][ks*32 + kbase].
__device__ __forceinline__ v16bf frag_ld_a(const bf16* p) {
  v8bf lo = *(const v8bf*)(p);
  v8bf hi = *(const v8bf*)(p + 16);
  return __builtin_shufflevector(lo, hi, 0,1,2,3,4,5,6,7,8,9,10,11,12,13,14,15);
}
// Same A pattern, converting from f32 source (LDS-staged x); vector b128 loads.
__device__ __forceinline__ v16bf frag_ld_a_f32(const float* p) {
  v4f f0 = *(const v4f*)(p);
  v4f f1 = *(const v4f*)(p + 4);
  v4f f2 = *(const v4f*)(p + 16);
  v4f f3 = *(const v4f*)(p + 20);
  v16bf a;
#pragma unroll
  for (int i = 0; i < 4; ++i) {
    a[i]      = (bf16)f0[i];
    a[i + 4]  = (bf16)f1[i];
    a[i + 8]  = (bf16)f2[i];
    a[i + 12] = (bf16)f3[i];
  }
  return a;
}
// B operand (32x16 bf16): lane holds K = kbase..kbase+7 and 16+kbase..16+kbase+7;
// caller passes &Wt_rowmajor[n][ks*32 + 8*(lane/16)]  (B[k][n] = W[n][k]).
__device__ __forceinline__ v16bf frag_ld_b(const bf16* p) {
  v8bf lo = *(const v8bf*)(p);
  v8bf hi = *(const v8bf*)(p + 16);
  return __builtin_shufflevector(lo, hi, 0,1,2,3,4,5,6,7,8,9,10,11,12,13,14,15);
}

// ---------- f32 -> bf16 weight conversion ------------------------------------
__global__ void cvt_f32_bf16(const float* __restrict__ src,
                             bf16* __restrict__ dst, int n) {
  int i = blockIdx.x * blockDim.x + threadIdx.x;
  if (i < n) dst[i] = (bf16)src[i];
}

// ---------- fused MHA: one workgroup per 64-token group, one wave per head ---
__global__ __launch_bounds__(256) void mha_fused(
    const float* __restrict__ x,     // [N][256] f32
    const bf16* __restrict__ wqkv,   // [768][256] bf16
    const bf16* __restrict__ wout,   // [256][256] bf16
    const float* __restrict__ bout,  // [256]
    float* __restrict__ out)         // [N][256] f32
{
  extern __shared__ bf16 smem[];
  // byte map (scalar-stored regions kept < 64 KB for DS immediate offsets):
  //  [0,32K)   Q  (later O)
  //  [32K,64K) K  (later attn staging)
  //  [64K,96K) V^T (packed stores only)
  //  [96K,160K) x tile f32 (async DMA + b128 reads only)
  bf16*  sQ   = smem;                   // [8][64][32]
  bf16*  sO   = smem;                   // [64][256] after phase-2 preload
  bf16*  sK   = smem + 16384;           // [8][64][32]
  bf16*  sAttn= smem + 16384;           // [8][16][64] aliases dead K
  bf16*  sVt  = smem + 32768;           // [256][64]  feature-major V^T
  float* sXf  = (float*)(smem + 49152); // [64][256] f32

  const int g     = blockIdx.x;
  const int tid   = threadIdx.x;
  const int wave  = tid >> 5;     // == head index
  const int lane  = tid & 31;
  const int lh    = lane & 15;
  const int khalf = lane >> 4;    // 0 or 1
  const long row0 = (long)g * GROUP;

  // ---------------- Phase 0: async DMA of x tile into LDS (f32, 64 KB) ------
  {
    const float* xg = x + row0 * EMBED;
#pragma unroll
    for (int i = 0; i < 16; ++i) {
      const float* src = xg + tid * 4 + i * 1024;     // 16 B per lane per round
      const int    dst = 98304 + tid * 16 + i * 4096; // LDS byte offset
      asm volatile("global_load_async_to_lds_b128 %0, %1, off"
                   :: "v"(dst), "v"(src) : "memory");
    }
    // warm this wave's w_qkv rows while the DMA runs
#pragma unroll
    for (int tgt = 0; tgt < 3; ++tgt)
#pragma unroll
      for (int ntl = 0; ntl < 2; ++ntl)
        __builtin_prefetch(wqkv + (long)(tgt * 256 + wave * 32 + ntl * 16 + lh) * EMBED
                                + khalf * 128, 0, 1);
    asm volatile("s_wait_asynccnt 0" ::: "memory");
  }
  __syncthreads();

  // ---------------- Phase 1: QKV projection (this wave's head) --------------
  // 6 concurrent accumulator chains (Q/K/V x 2 column tiles); A-fragment is
  // software-pipelined (load/convert k+1 while WMMAs consume k) to kill the
  // WMMA->VALU WAR hazard nops.
  {
    const float* xrow_base = sXf + lh * EMBED + khalf * 8;
    const bf16* wrow[6];
#pragma unroll
    for (int j = 0; j < 6; ++j) {
      const int tgt = j >> 1, ntl = j & 1;
      wrow[j] = wqkv + (long)(tgt * 256 + wave * 32 + ntl * 16 + lh) * EMBED
                     + khalf * 8;
    }
#pragma unroll 1
    for (int mt = 0; mt < 4; ++mt) {
      const float* xrow = xrow_base + mt * 16 * EMBED;
      v8f acc[6];
#pragma unroll
      for (int j = 0; j < 6; ++j) acc[j] = zero8();
      v16bf afr = frag_ld_a_f32(xrow);
#pragma unroll
      for (int ks = 0; ks < 8; ++ks) {
        v16bf anext = afr;
        if (ks < 7) anext = frag_ld_a_f32(xrow + (ks + 1) * 32);
#pragma unroll
        for (int j = 0; j < 6; ++j)
          acc[j] = wmma_bf16(afr, frag_ld_b(wrow[j] + ks * 32), acc[j]);
        afr = anext;
      }
#pragma unroll
      for (int j = 0; j < 6; ++j) {
        const int tgt = j >> 1, ntl = j & 1;
        if (tgt < 2) {                 // Q,K row-major [64][32] per head
          bf16* dst = (tgt == 0 ? sQ : sK) + wave * 2048;
          const int col = ntl * 16 + lh;
#pragma unroll
          for (int v = 0; v < 8; ++v)
            dst[(mt * 16 + v + 8 * khalf) * HD + col] = (bf16)acc[j][v];
        } else {                       // V^T: token axis contiguous -> pack
          bf16* dst = sVt + (wave * 32 + ntl * 16 + lh) * GROUP
                          + mt * 16 + khalf * 8;
          v8bf pk;
#pragma unroll
          for (int v = 0; v < 8; ++v) pk[v] = (bf16)acc[j][v];
          *(v8bf*)dst = pk;            // one 16-byte DS store
        }
      }
    }
  }
  __syncthreads();

  // ---------------- Phase 2: attention for head == wave ---------------------
  {
    const bf16* qh = sQ + wave * 2048;
    const bf16* kh = sK + wave * 2048;
    const bf16* vt = sVt + wave * 32 * GROUP;
    bf16*       ah = sAttn + wave * 1024;      // [16][64] staging tile

    v16bf kfr[4], vfr[2][2], qfr[4];
#pragma unroll
    for (int nt = 0; nt < 4; ++nt)             // B of Q·K^T: rows of K
      kfr[nt] = frag_ld_b(kh + (nt * 16 + lh) * HD + khalf * 8);
#pragma unroll
    for (int ntv = 0; ntv < 2; ++ntv)          // B of attn·V: rows of V^T
#pragma unroll
      for (int ks = 0; ks < 2; ++ks)
        vfr[ntv][ks] = frag_ld_b(vt + (ntv * 16 + lh) * GROUP + ks * 32 + khalf * 8);
#pragma unroll
    for (int mt = 0; mt < 4; ++mt)
      qfr[mt] = frag_ld_a(qh + (mt * 16 + lh) * HD + khalf * 8);

    __syncthreads();   // all Q/K reads done; attn/O writes may begin

#pragma unroll
    for (int mt = 0; mt < 4; ++mt) {
      v8f e[4];
#pragma unroll
      for (int nt = 0; nt < 4; ++nt)
        e[nt] = wmma_bf16(qfr[mt], kfr[nt], zero8());

      // softmax over 64 columns; row = mt*16 + v + 8*khalf, cols across lanes
#pragma unroll
      for (int v = 0; v < 8; ++v) {
        float mx = fmaxf(fmaxf(e[0][v], e[1][v]), fmaxf(e[2][v], e[3][v]));
#pragma unroll
        for (int off = 1; off < 16; off <<= 1)
          mx = fmaxf(mx, __shfl_xor(mx, off, 32));
        float s = 0.f;
#pragma unroll
        for (int nt = 0; nt < 4; ++nt) {
          float p = __expf((e[nt][v] - mx) * SM_SCALE);
          e[nt][v] = p;
          s += p;
        }
#pragma unroll
        for (int off = 1; off < 16; off <<= 1)
          s += __shfl_xor(s, off, 32);
        float inv = __builtin_amdgcn_rcpf(s);
#pragma unroll
        for (int nt = 0; nt < 4; ++nt) e[nt][v] *= inv;
      }

      // D-layout -> A-layout via per-head LDS staging (rows of this m-tile)
#pragma unroll
      for (int nt = 0; nt < 4; ++nt) {
        const int col = nt * 16 + lh;
#pragma unroll
        for (int v = 0; v < 8; ++v)
          ah[(v + 8 * khalf) * GROUP + col] = (bf16)e[nt][v];
      }

      v8f oacc0 = zero8(), oacc1 = zero8();
#pragma unroll
      for (int ks = 0; ks < 2; ++ks) {
        v16bf afr = frag_ld_a(ah + lh * GROUP + ks * 32 + khalf * 8);
        oacc0 = wmma_bf16(afr, vfr[0][ks], oacc0);
        oacc1 = wmma_bf16(afr, vfr[1][ks], oacc1);
      }

      // write O tile (64x256 bf16 row-major into sQ region, offsets < 32K)
#pragma unroll
      for (int v = 0; v < 8; ++v) {
        const int m = mt * 16 + v + 8 * khalf;
        sO[m * EMBED + wave * 32 + lh]      = (bf16)oacc0[v];
        sO[m * EMBED + wave * 32 + 16 + lh] = (bf16)oacc1[v];
      }
    }
  }
  __syncthreads();

  // ---------------- Phase 3: output projection (wave owns 32 out cols) ------
  // Two interleaved accumulator chains; A-fragment pipelined one step ahead.
  {
    const int n0 = wave * 32;
    float* og = out + row0 * EMBED;
    const bf16* wrow0 = wout + (long)(n0 + lh) * EMBED + khalf * 8;
    const bf16* wrow1 = wout + (long)(n0 + 16 + lh) * EMBED + khalf * 8;
    const float bias0 = bout[n0 + lh];
    const float bias1 = bout[n0 + 16 + lh];
#pragma unroll 1
    for (int mt = 0; mt < 4; ++mt) {
      const bf16* orow = sO + (mt * 16 + lh) * EMBED + khalf * 8;
      v8f acc0 = zero8(), acc1 = zero8();
      v16bf afr = frag_ld_a(orow);
#pragma unroll
      for (int ks = 0; ks < 8; ++ks) {
        v16bf anext = afr;
        if (ks < 7) anext = frag_ld_a(orow + (ks + 1) * 32);
        acc0 = wmma_bf16(afr, frag_ld_b(wrow0 + ks * 32), acc0);
        acc1 = wmma_bf16(afr, frag_ld_b(wrow1 + ks * 32), acc1);
        afr = anext;
      }
#pragma unroll
      for (int v = 0; v < 8; ++v) {
        const int m = mt * 16 + v + 8 * khalf;
        og[(long)m * EMBED + n0 + lh]      = acc0[v] + bias0;
        og[(long)m * EMBED + n0 + 16 + lh] = acc1[v] + bias1;
      }
    }
  }
}

extern "C" void kernel_launch(void* const* d_in, const int* in_sizes, int n_in,
                              void* d_out, int out_size, void* d_ws, size_t ws_size,
                              hipStream_t stream) {
  const float* x    = (const float*)d_in[0];
  const float* wqkv = (const float*)d_in[1];
  const float* wout = (const float*)d_in[2];
  const float* bout = (const float*)d_in[3];
  float* out = (float*)d_out;

  const int N = in_sizes[0] / EMBED;          // 131072
  const int ngroups = N / GROUP;              // 2048

  bf16* wqkv_bf = (bf16*)d_ws;                // 768*256 bf16
  bf16* wout_bf = wqkv_bf + 768 * EMBED;      // 256*256 bf16

  cvt_f32_bf16<<<(768 * EMBED + 255) / 256, 256, 0, stream>>>(wqkv, wqkv_bf, 768 * EMBED);
  cvt_f32_bf16<<<(EMBED * EMBED + 255) / 256, 256, 0, stream>>>(wout, wout_bf, EMBED * EMBED);

  const size_t lds_bytes = 163840;            // 160 KB: Q+K+Vt (96K) + x f32 (64K)
  mha_fused<<<ngroups, 256, lds_bytes, stream>>>(x, wqkv_bf, wout_bf, bout, out);
}
